// HMPGNNplus_11278584119458
// MI455X (gfx1250) — compile-verified, matrix operand order checked by reading, not run on hardware
//
#include <hip/hip_runtime.h>
#include <hip/hip_bf16.h>

typedef _Float16 v4h  __attribute__((ext_vector_type(4)));
typedef _Float16 v8h  __attribute__((ext_vector_type(8)));
typedef _Float16 v16h __attribute__((ext_vector_type(16)));
typedef float    v8f  __attribute__((ext_vector_type(8)));

#define D_IN   128
#define D_OUT  64
#define TILE_M 32
#define ASTR   136   // 128 + 8 halves padding (272B row stride -> 4-bank shift)
#define WSTR   136

__device__ __forceinline__ float wave_sum32(float v) {
    #pragma unroll
    for (int off = 16; off > 0; off >>= 1) v += __shfl_xor(v, off, 32);
    return v;
}

// ---------------- bias expmap0 (tiny, 1 block x 64 threads) ----------------
__global__ void bias_expmap_kernel(const float* __restrict__ bias,
                                   const float* __restrict__ cp,
                                   float* __restrict__ hypb) {
    __shared__ float sb[64];
    int t = threadIdx.x;
    float v = bias[t];
    sb[t] = v * v;
    __syncthreads();
    for (int off = 32; off > 0; off >>= 1) {
        if (t < off) sb[t] += sb[t + off];
        __syncthreads();
    }
    float c  = cp[0];
    float sc = sqrtf(c);
    float un = fmaxf(sqrtf(sb[0]), 1e-15f);
    hypb[t] = tanhf(sc * un) * v / (sc * un);
}

// ---------------- GEMM: res = x @ W^T via v_wmma_f32_16x16x32_f16 ----------
__global__ __launch_bounds__(256) void gemm_wmma_kernel(
        const float* __restrict__ X, const float* __restrict__ W,
        float* __restrict__ res, int N) {
    __shared__ _Float16 Alds[TILE_M * ASTR];
    __shared__ _Float16 Wlds[D_OUT * WSTR];
    const int tid  = threadIdx.x;
    const int row0 = blockIdx.x * TILE_M;
    const bool full = (row0 + TILE_M) <= N;   // 3124/3125 blocks: branchless path

    // Stage full 64x128 weight as f16 (2048 float4 loads across 256 threads)
    for (int i = tid; i < (D_OUT * D_IN) / 4; i += 256) {
        int r = i >> 5, c4 = i & 31;
        float4 f = ((const float4*)W)[r * 32 + c4];
        v4h h = { (_Float16)f.x, (_Float16)f.y, (_Float16)f.z, (_Float16)f.w };
        *(v4h*)&Wlds[r * WSTR + c4 * 4] = h;
    }
    // Stage 32x128 x panel as f16
    if (full) {
        #pragma unroll
        for (int i = tid; i < (TILE_M * D_IN) / 4; i += 256) {
            int r = i >> 5, c4 = i & 31;
            float4 f = ((const float4*)X)[(size_t)(row0 + r) * 32 + c4];
            v4h h = { (_Float16)f.x, (_Float16)f.y, (_Float16)f.z, (_Float16)f.w };
            *(v4h*)&Alds[r * ASTR + c4 * 4] = h;
        }
    } else {
        for (int i = tid; i < (TILE_M * D_IN) / 4; i += 256) {
            int r = i >> 5, c4 = i & 31;
            int grow = row0 + r;
            float4 f = {0.f, 0.f, 0.f, 0.f};
            if (grow < N) f = ((const float4*)X)[(size_t)grow * 32 + c4];
            v4h h = { (_Float16)f.x, (_Float16)f.y, (_Float16)f.z, (_Float16)f.w };
            *(v4h*)&Alds[r * ASTR + c4 * 4] = h;
        }
    }
    __syncthreads();

    const int lane = tid & 31, wv = tid >> 5;
    const int rt  = (wv >> 2) << 4;   // row tile: 0 or 16
    const int ct  = (wv & 3) << 4;    // col tile: 0,16,32,48
    const int g   = lane >> 4;        // lane group
    const int idx = lane & 15;

    const _Float16* arow = &Alds[(rt + idx) * ASTR];
    const _Float16* brow = &Wlds[(ct + idx) * WSTR];

    v8f acc = {};
    #pragma unroll
    for (int kb = 0; kb < D_IN; kb += 32) {
        // Per-lane A/B fragments: K in [kb+8g, kb+8g+8) and [kb+16+8g, +8)
        union { v16h v; v8h h[2]; } a, b;
        a.h[0] = *(const v8h*)&arow[kb + 8 * g];
        a.h[1] = *(const v8h*)&arow[kb + 16 + 8 * g];
        b.h[0] = *(const v8h*)&brow[kb + 8 * g];
        b.h[1] = *(const v8h*)&brow[kb + 16 + 8 * g];
        acc = __builtin_amdgcn_wmma_f32_16x16x32_f16(
            false, a.v, false, b.v, (short)0, acc, false, false);
    }

    // D layout: VGPR r -> row M = r + 8g, col N = idx
    float* dbase = res + (size_t)(row0 + rt + 8 * g) * D_OUT + ct + idx;
    if (full) {
        #pragma unroll
        for (int r = 0; r < 8; r++) dbase[(size_t)r * D_OUT] = acc[r];
    } else {
        #pragma unroll
        for (int r = 0; r < 8; r++) {
            int grow = row0 + rt + r + 8 * g;
            if (grow < N) res[(size_t)grow * D_OUT + ct + idx] = acc[r];
        }
    }
}

// ------------- per-node: proj -> mobius_add -> proj -> p2k -> lamb ---------
__global__ __launch_bounds__(256) void node_transform_kernel(
        const float* __restrict__ res, const float* __restrict__ hypb,
        const float* __restrict__ cp, float* __restrict__ s,
        float* __restrict__ lamb, int N) {
    const int lane = threadIdx.x & 31, wv = threadIdx.x >> 5;
    const int node = blockIdx.x * 8 + wv;
    if (node >= N) return;
    const float c  = cp[0];
    const float sc = sqrtf(c);
    const float maxn = (1.0f - 4e-3f) / sc;

    const float* r = res + (size_t)node * D_OUT;
    float x0 = r[lane], x1 = r[lane + 32];

    // proj
    float n2 = wave_sum32(x0 * x0 + x1 * x1);
    float nrm = fmaxf(sqrtf(n2), 1e-15f);
    float scl = (nrm > maxn) ? (maxn / nrm) : 1.0f;
    x0 *= scl; x1 *= scl;
    float x2 = n2 * scl * scl;

    // mobius_add with hyp_bias
    float y0 = hypb[lane], y1 = hypb[lane + 32];
    float y2 = wave_sum32(y0 * y0 + y1 * y1);
    float xy = wave_sum32(x0 * y0 + x1 * y1);
    float k1 = 1.0f + 2.0f * c * xy + c * y2;
    float k2 = 1.0f - c * x2;
    float den = fmaxf(1.0f + 2.0f * c * xy + c * c * x2 * y2, 1e-15f);
    float m0 = (k1 * x0 + k2 * y0) / den;
    float m1 = (k1 * x1 + k2 * y1) / den;

    // proj
    float m2 = wave_sum32(m0 * m0 + m1 * m1);
    nrm = fmaxf(sqrtf(m2), 1e-15f);
    scl = (nrm > maxn) ? (maxn / nrm) : 1.0f;
    m0 *= scl; m1 *= scl; m2 *= scl * scl;

    // p2k
    float pden = 1.0f + c * m2;
    float s0 = 2.0f * m0 / pden, s1 = 2.0f * m1 / pden;

    // lorenz factor (reference uses default c=1.0 here)
    float s2 = wave_sum32(s0 * s0 + s1 * s1);
    float lam = rsqrtf(fmaxf(1.0f - s2, 1e-15f));

    s[(size_t)node * D_OUT + lane]      = s0;
    s[(size_t)node * D_OUT + lane + 32] = s1;
    if (lane == 0) lamb[node] = lam;
}

// ---------------- degree: init (self loop), edges, rsqrt -------------------
__global__ void deg_init_kernel(float* __restrict__ deg, int N) {
    int i = blockIdx.x * blockDim.x + threadIdx.x;
    if (i < N) deg[i] = 1.0f;
}
__global__ void deg_edge_kernel(const long long* __restrict__ ei,
                                float* __restrict__ deg, int E) {
    int e = blockIdx.x * blockDim.x + threadIdx.x;
    if (e < E) atomicAdd(&deg[(int)ei[e]], 1.0f);
}
__global__ void deg_fin_kernel(float* __restrict__ deg, int N) {
    int i = blockIdx.x * blockDim.x + threadIdx.x;
    if (i < N) { float d = deg[i]; deg[i] = d > 0.0f ? rsqrtf(d) : 0.0f; }
}

// ---------------- edge aggregation: one wave per edge ----------------------
__global__ __launch_bounds__(256) void aggregate_kernel(
        const long long* __restrict__ ei, const float* __restrict__ dis,
        const float* __restrict__ lamb, const float* __restrict__ s,
        float* __restrict__ acc, float* __restrict__ tmp, int E) {
    const int lane = threadIdx.x & 31, wv = threadIdx.x >> 5;
    const int e = blockIdx.x * 8 + wv;
    if (e >= E) return;
    const int r = (int)ei[e];
    const int cI = (int)ei[(size_t)E + e];
    const float w = dis[r] * dis[cI] * lamb[cI];
    const size_t cb = (size_t)cI * D_OUT, rb = (size_t)r * D_OUT;
    atomicAdd(&acc[rb + lane],      w * s[cb + lane]);
    atomicAdd(&acc[rb + lane + 32], w * s[cb + lane + 32]);
    if (lane == 0) atomicAdd(&tmp[r], w);
}

// ---------------- finalize: self loop + divide + k2p + leaky_relu ----------
__global__ __launch_bounds__(256) void finalize_kernel(
        const float* __restrict__ acc, const float* __restrict__ tmp,
        const float* __restrict__ s, const float* __restrict__ dis,
        const float* __restrict__ lamb, const float* __restrict__ cp,
        float* __restrict__ out, int N) {
    const int lane = threadIdx.x & 31, wv = threadIdx.x >> 5;
    const int node = blockIdx.x * 8 + wv;
    if (node >= N) return;
    const float c = cp[0];
    const float di = dis[node];
    const float wself = di * di * lamb[node];
    const size_t b = (size_t)node * D_OUT;
    float a0 = acc[b + lane]      + wself * s[b + lane];
    float a1 = acc[b + lane + 32] + wself * s[b + lane + 32];
    float t = tmp[node] + wself;
    float so0 = a0 / t, so1 = a1 / t;
    float s2 = wave_sum32(so0 * so0 + so1 * so1);
    float kden = 1.0f + sqrtf(fmaxf(1.0f - c * s2, 1e-15f));
    float o0 = so0 / kden, o1 = so1 / kden;
    out[b + lane]      = o0 > 0.0f ? o0 : 0.01f * o0;
    out[b + lane + 32] = o1 > 0.0f ? o1 : 0.01f * o1;
}

extern "C" void kernel_launch(void* const* d_in, const int* in_sizes, int n_in,
                              void* d_out, int out_size, void* d_ws, size_t ws_size,
                              hipStream_t stream) {
    const float*     x    = (const float*)d_in[0];      // [N,128]
    const float*     w    = (const float*)d_in[1];      // [64,128]
    const float*     bias = (const float*)d_in[2];      // [1,64]
    const float*     cptr = (const float*)d_in[3];      // scalar
    const long long* ei   = (const long long*)d_in[4];  // [2,E] int64

    const int N = in_sizes[0] / D_IN;
    const int E = in_sizes[4] / 2;
    float* out = (float*)d_out;

    // workspace partition (256B aligned blocks)
    char* base = (char*)d_ws;
    size_t off = 0;
    auto take = [&](size_t bytes) -> char* {
        char* p = base + off;
        off += (bytes + 255) & ~(size_t)255;
        return p;
    };
    float* res  = (float*)take((size_t)N * D_OUT * sizeof(float));
    float* sbuf = (float*)take((size_t)N * D_OUT * sizeof(float));
    float* acc  = (float*)take((size_t)N * D_OUT * sizeof(float));
    float* lamb = (float*)take((size_t)N * sizeof(float));
    float* deg  = (float*)take((size_t)N * sizeof(float));  // becomes dis in-place
    float* tmp  = (float*)take((size_t)N * sizeof(float));
    float* hypb = (float*)take(D_OUT * sizeof(float));
    (void)ws_size; (void)n_in; (void)out_size;

    hipMemsetAsync(acc, 0, (size_t)N * D_OUT * sizeof(float), stream);
    hipMemsetAsync(tmp, 0, (size_t)N * sizeof(float), stream);

    bias_expmap_kernel<<<1, 64, 0, stream>>>(bias, cptr, hypb);

    gemm_wmma_kernel<<<(N + TILE_M - 1) / TILE_M, 256, 0, stream>>>(x, w, res, N);

    node_transform_kernel<<<(N + 7) / 8, 256, 0, stream>>>(res, hypb, cptr,
                                                           sbuf, lamb, N);

    deg_init_kernel<<<(N + 255) / 256, 256, 0, stream>>>(deg, N);
    deg_edge_kernel<<<(E + 255) / 256, 256, 0, stream>>>(ei, deg, E);
    deg_fin_kernel<<<(N + 255) / 256, 256, 0, stream>>>(deg, N);

    aggregate_kernel<<<(E + 7) / 8, 256, 0, stream>>>(ei, deg, lamb, sbuf,
                                                      acc, tmp, E);

    finalize_kernel<<<(N + 7) / 8, 256, 0, stream>>>(acc, tmp, sbuf, deg, lamb,
                                                     cptr, out, N);
}
